// EmformerBlock_62766652063949
// MI455X (gfx1250) — matrix operand order repaired and checked by reading, not compiled, not run on hardware
//
#include <hip/hip_runtime.h>
#include <hip/hip_bf16.h>
#include <stdint.h>

// ---------------------------------------------------------------------------
// Emformer block forward for MI455X (gfx1250, wave32, WMMA).
// GEMMs: v_wmma_f32_16x16x32_f16, both LDS tiles staged with async
// global->LDS (weights pre-transposed so the copy is linear).
// Attention: transposed S^T/O^T formulation, zero LDS, zero barriers.
// ---------------------------------------------------------------------------

typedef _Float16 half8  __attribute__((ext_vector_type(8)));
typedef _Float16 half4t __attribute__((ext_vector_type(4)));
typedef _Float16 v16h   __attribute__((ext_vector_type(16)));
typedef float    v8f    __attribute__((ext_vector_type(8)));
typedef int      v4i_   __attribute__((ext_vector_type(4)));
typedef unsigned v8u    __attribute__((ext_vector_type(8)));

#define B_      2
#define S_      2048
#define D_      512
#define H_      8
#define DK_     64
#define DFF_    2048
#define CHUNK_  128
#define LEFT_   512
#define RIGHT_  64
#define ROWS_   (B_ * S_)          // 4096
#define QKVL_   (3 * D_)           // fused q|k|v row stride = 1536
#define EPS_    1e-3f

#if defined(__gfx1250__) &&                                                   \
    __has_builtin(__builtin_amdgcn_global_load_async_to_lds_b128) &&          \
    __has_builtin(__builtin_amdgcn_s_wait_asynccnt)
#define USE_ASYNC 1
#else
#define USE_ASYNC 0
#endif

static __device__ __forceinline__ v8f wmma_f16(v16h a, v16h b, v8f c) {
  return __builtin_amdgcn_wmma_f32_16x16x32_f16(false, a, false, b,
                                                (short)0, c, false, false);
}
static __device__ __forceinline__ v16h combine8(half8 lo, half8 hi) {
  return __builtin_shufflevector(lo, hi, 0,1,2,3,4,5,6,7,8,9,10,11,12,13,14,15);
}

// Exchange with lane^16 (cross half-wave), one VALU op on gfx12.
static __device__ __forceinline__ unsigned xhalf_u(unsigned v) {
#if __has_builtin(__builtin_amdgcn_permlanex16)
  return (unsigned)__builtin_amdgcn_permlanex16(
      (int)v, (int)v, 0x76543210, 0xFEDCBA98, false, false);
#else
  return (unsigned)__shfl_xor((int)v, 16, 32);
#endif
}
static __device__ __forceinline__ float xhalf_f(float v) {
  return __builtin_bit_cast(float, xhalf_u(__builtin_bit_cast(unsigned, v)));
}

static __device__ __forceinline__ float max8(const v8f& x) {
  float a = fmaxf(fmaxf(x[0], x[1]), fmaxf(x[2], x[3]));
  float b = fmaxf(fmaxf(x[4], x[5]), fmaxf(x[6], x[7]));
  return fmaxf(a, b);
}
static __device__ __forceinline__ float sum8(const v8f& x) {
  return ((x[0] + x[1]) + (x[2] + x[3])) + ((x[4] + x[5]) + (x[6] + x[7]));
}

#if USE_ASYNC
static __device__ __forceinline__ void async_cp16(const void* g, void* l) {
  auto* gp = reinterpret_cast<v4i_ __attribute__((address_space(1)))*>(
      reinterpret_cast<uintptr_t>(g));
  auto* lp = reinterpret_cast<v4i_ __attribute__((address_space(3)))*>(
      reinterpret_cast<uintptr_t>(l));
  __builtin_amdgcn_global_load_async_to_lds_b128(gp, lp, 0, 0);
}
#endif

// ---------------------------------------------------------------------------
// f32 -> f16 conversion, TRANSPOSED destination: in is [K,N] row-major f32,
// out is W^T [N,Kdim] f16. n4 = total float4 units, N4 = N/4.
// ---------------------------------------------------------------------------
__global__ __launch_bounds__(256) void cvt_f32_f16_T(const float* __restrict__ in,
                                                     _Float16* __restrict__ out,
                                                     int n4, int N4, int Kdim) {
  int i = blockIdx.x * 256 + threadIdx.x;
  if (i < n4) {
    float4 v = ((const float4*)in)[i];
    int k  = i / N4;
    int n0 = (i - k * N4) * 4;
    out[(size_t)(n0 + 0) * Kdim + k] = (_Float16)v.x;
    out[(size_t)(n0 + 1) * Kdim + k] = (_Float16)v.y;
    out[(size_t)(n0 + 2) * Kdim + k] = (_Float16)v.z;
    out[(size_t)(n0 + 3) * Kdim + k] = (_Float16)v.w;
  }
}

__global__ __launch_bounds__(256) void pack_bias3(const float* __restrict__ a,
                                                  const float* __restrict__ b,
                                                  const float* __restrict__ c,
                                                  float* __restrict__ out) {
  int i = blockIdx.x * 256 + threadIdx.x;   // 0..1535
  float v = (i < D_) ? a[i] : (i < 2 * D_) ? b[i - D_] : c[i - 2 * D_];
  out[i] = v;
}

// ---------------------------------------------------------------------------
// LayerNorm over D=512, one block (128 threads) per row.
// MODE 0: y = LN(A); store y32 + y16                 (input LN)
// MODE 1: pre = A + Badd; store pre32; y16 = LN(pre) (post-attention LN)
// MODE 2: y = LN(A); store y32 only                  (final LN)
// ---------------------------------------------------------------------------
template <int MODE>
__global__ __launch_bounds__(128) void ln_kernel(const float* __restrict__ A,
                                                 const float* __restrict__ Badd,
                                                 const float* __restrict__ g,
                                                 const float* __restrict__ beta,
                                                 float* __restrict__ preOut,
                                                 float* __restrict__ y32,
                                                 _Float16* __restrict__ y16) {
  __shared__ float r1[128];
  __shared__ float r2[128];
  const int row = blockIdx.x;
  const int tid = threadIdx.x;
  const size_t base = (size_t)row * D_;

  float4 x = ((const float4*)(A + base))[tid];
  if constexpr (MODE == 1) {
    float4 y = ((const float4*)(Badd + base))[tid];
    x.x += y.x; x.y += y.y; x.z += y.z; x.w += y.w;
    ((float4*)(preOut + base))[tid] = x;
  }
  float s1 = x.x + x.y + x.z + x.w;
  float s2 = x.x * x.x + x.y * x.y + x.z * x.z + x.w * x.w;
  r1[tid] = s1; r2[tid] = s2;
  __syncthreads();
  for (int off = 64; off > 0; off >>= 1) {
    if (tid < off) { r1[tid] += r1[tid + off]; r2[tid] += r2[tid + off]; }
    __syncthreads();
  }
  const float mu  = r1[0] * (1.0f / D_);
  const float var = r2[0] * (1.0f / D_) - mu * mu;
  const float inv = rsqrtf(var + EPS_);

  float4 gg = ((const float4*)g)[tid];
  float4 bb = ((const float4*)beta)[tid];
  float o0 = (x.x - mu) * inv * gg.x + bb.x;
  float o1 = (x.y - mu) * inv * gg.y + bb.y;
  float o2 = (x.z - mu) * inv * gg.z + bb.z;
  float o3 = (x.w - mu) * inv * gg.w + bb.w;
  if constexpr (MODE != 1) {
    float4 o; o.x = o0; o.y = o1; o.z = o2; o.w = o3;
    ((float4*)(y32 + base))[tid] = o;
  }
  if constexpr (MODE != 2) {
    half4t h = {(_Float16)o0, (_Float16)o1, (_Float16)o2, (_Float16)o3};
    ((half4t*)(y16 + base))[tid] = h;
  }
}

// ---------------------------------------------------------------------------
// Tiled WMMA GEMM:  C[M,N] = A[M,K](f16) * W^T[N,K](f16) + bias (+ residual)
// Block tile 128x128, 8 waves x (64x32), BK=32, LDS double-buffered.
// Both tiles are linear copies (W is pre-transposed) staged with async
// global->LDS b128.
// MODE 0: out16 = f16(val)
// MODE 1: out32 = val + residual
// MODE 2: QKV epilogue: cols <1024 -> out16 (q|k); cols >=1024 -> vtg
//         (V transposed to [B*H*DK, S] f16 for the attention kernel).
// ---------------------------------------------------------------------------
template <int MODE>
__global__ __launch_bounds__(256) void gemm_wmma(const _Float16* __restrict__ A,
                                                 const _Float16* __restrict__ WT,
                                                 const float* __restrict__ bias,
                                                 _Float16* __restrict__ out16,
                                                 float* __restrict__ out32,
                                                 const float* __restrict__ residual,
                                                 _Float16* __restrict__ vtg,
                                                 int M, int N, int K) {
  __shared__ _Float16 sA[2][128 * 40];   // [row][k], stride 40
  __shared__ _Float16 sB[2][128 * 40];   // [col][k], stride 40

  const int nb   = N >> 7;
  const int bm   = (blockIdx.x / nb) << 7;
  const int bn   = (blockIdx.x % nb) << 7;
  const int tid  = threadIdx.x;
  const int w    = tid >> 5;
  const int lane = tid & 31;
  const int ln16 = lane & 15;
  const int hb   = lane >> 4;
  const int wm   = (w & 1) * 64;
  const int wn   = (w >> 1) * 32;

  const int cr = tid >> 1, cc = (tid & 1) * 16;   // tile copy coords

  auto stage = [&](int k0, int buf) {
    const _Float16* ga = A  + (size_t)(bm + cr) * K + k0 + cc;
    const _Float16* gb = WT + (size_t)(bn + cr) * K + k0 + cc;
    _Float16* la = &sA[buf][cr * 40 + cc];
    _Float16* lb = &sB[buf][cr * 40 + cc];
#if USE_ASYNC
    async_cp16(ga, la); async_cp16(ga + 8, la + 8);
    async_cp16(gb, lb); async_cp16(gb + 8, lb + 8);
#else
    *(v16h*)la = *(const v16h*)ga;
    *(v16h*)lb = *(const v16h*)gb;
#endif
  };

  v8f acc[4][2];
#pragma unroll
  for (int mt = 0; mt < 4; ++mt)
#pragma unroll
    for (int nt = 0; nt < 2; ++nt) acc[mt][nt] = (v8f){};

  const int nk = K >> 5;
  stage(0, 0);
  for (int kt = 0; kt < nk; ++kt) {
    const int cur = kt & 1;
#if USE_ASYNC
    __builtin_amdgcn_s_wait_asynccnt(0);
#endif
    __syncthreads();
    if (kt + 1 < nk) stage((kt + 1) << 5, cur ^ 1);

    v16h af[4], bf[2];
#pragma unroll
    for (int mt = 0; mt < 4; ++mt) {
      const _Float16* p = &sA[cur][(wm + mt * 16 + ln16) * 40 + hb * 16];
      af[mt] = combine8(*(const half8*)p, *(const half8*)(p + 8));
    }
#pragma unroll
    for (int nt = 0; nt < 2; ++nt)
      bf[nt] = *(const v16h*)&sB[cur][(wn + nt * 16 + ln16) * 40 + hb * 16];
#pragma unroll
    for (int mt = 0; mt < 4; ++mt)
#pragma unroll
      for (int nt = 0; nt < 2; ++nt)
        acc[mt][nt] = wmma_f16(af[mt], bf[nt], acc[mt][nt]);
  }

  // Branch-free epilogue. C layout: element r -> row r+8*hb, col = lane%16.
#pragma unroll
  for (int nt = 0; nt < 2; ++nt) {
    const int col = bn + wn + nt * 16 + ln16;
    const float bb = bias[col];
#pragma unroll
    for (int mt = 0; mt < 4; ++mt) {
#pragma unroll
      for (int r = 0; r < 8; ++r) {
        const int row = bm + wm + mt * 16 + r + 8 * hb;
        const size_t idx = (size_t)row * N + col;
        const float val = acc[mt][nt][r] + bb;
        if constexpr (MODE == 0) {
          out16[idx] = (_Float16)val;
        } else if constexpr (MODE == 1) {
          out32[idx] = val + residual[idx];
        } else {
          if (col < 2 * D_) {                       // q|k sections
            out16[idx] = (_Float16)val;
          } else {                                  // v -> transposed buffer
            const int dk = col - 2 * D_;            // h*64 + dk_in_head
            const int bidx = row >> 11;             // row / S_
            const int s = row & (S_ - 1);
            vtg[((size_t)(bidx * H_ * DK_ + dk)) * S_ + s] = (_Float16)val;
          }
        }
      }
    }
  }
}

// ---------------------------------------------------------------------------
// Windowed (Emformer) flash attention, transposed formulation, LDS-free.
//   S^T = K * Q^T   (C layout: col = query = lane%16, rows = keys)
//   O^T = V^T * P^T (C layout: col = query, rows = dk)
// Per-query softmax state is a per-lane scalar; reductions are in-lane + one
// permlanex16. All operands stream from global (L2-resident).
// qkv f16 [B,S,1536]: q cols 0.., k cols 512..; vtg f16 [B*H*DK, S].
// ---------------------------------------------------------------------------
__global__ __launch_bounds__(256) void attn_wmma(const _Float16* __restrict__ qkv,
                                                 const _Float16* __restrict__ vtg,
                                                 float* __restrict__ attn) {
  const int blk = blockIdx.x;             // b*128 + h*16 + ci
  const int ci  = blk & 15;
  const int h   = (blk >> 4) & 7;
  const int b   = blk >> 7;
  int lo = ci * CHUNK_ - LEFT_;           if (lo < 0) lo = 0;
  int hi = (ci + 1) * CHUNK_ + RIGHT_;    if (hi > S_) hi = S_;

  const int tid  = threadIdx.x;
  const int w    = tid >> 5;
  const int lane = tid & 31;
  const int ln16 = lane & 15;
  const int hb   = lane >> 4;

  // Q^T B-fragments: lane col = query; element e: dk = ks*32 + hb*16 + e.
  const int q = ci * CHUNK_ + w * 16 + ln16;
  const _Float16* qp = qkv + ((size_t)b * S_ + q) * QKVL_ + h * DK_;
  const v16h qb0 = *(const v16h*)(qp + hb * 16);
  const v16h qb1 = *(const v16h*)(qp + 32 + hb * 16);

  // V^T row bases: dk rows t*16+ln16 of head (b,h) in vtg.
  const _Float16* vrow[4];
#pragma unroll
  for (int t = 0; t < 4; ++t)
    vrow[t] = vtg + ((size_t)(b * H_ * DK_ + h * DK_ + t * 16 + ln16)) * S_;

  v8f acc[4];
#pragma unroll
  for (int t = 0; t < 4; ++t) acc[t] = (v8f){};
  float m = -3.0e38f, l = 0.0f;

  for (int kv = lo; kv < hi; kv += 32) {
    // S^T = K * Q^T: A-frags are K rows (keys) loaded straight from global.
    v8f st0 = (v8f){}, st1 = (v8f){};
#pragma unroll
    for (int ks = 0; ks < 2; ++ks) {
      const _Float16* kb = qkv + ((size_t)b * S_ + kv + ln16) * QKVL_ + D_ +
                           h * DK_ + ks * 32 + hb * 16;
      v16h ka0 = combine8(*(const half8*)kb, *(const half8*)(kb + 8));
      const _Float16* kb1 = kb + 16 * QKVL_;
      v16h ka1 = combine8(*(const half8*)kb1, *(const half8*)(kb1 + 8));
      const v16h qf = ks ? qb1 : qb0;
      st0 = wmma_f16(ka0, qf, st0);   // keys kv+0..15
      st1 = wmma_f16(ka1, qf, st1);   // keys kv+16..31
    }
#pragma unroll
    for (int r = 0; r < 8; ++r) { st0[r] *= 0.125f; st1[r] *= 0.125f; }

    // Per-query online softmax: in-lane over 16 scores + one lane^16 swap.
    float tm = fmaxf(max8(st0), max8(st1));
    tm = fmaxf(tm, xhalf_f(tm));
    const float nm    = fmaxf(m, tm);
    const float alpha = __expf(m - nm);
#pragma unroll
    for (int r = 0; r < 8; ++r) {
      st0[r] = __expf(st0[r] - nm);
      st1[r] = __expf(st1[r] - nm);
    }
    float ps = sum8(st0) + sum8(st1);
    ps += xhalf_f(ps);
    l = l * alpha + ps;
    m = nm;
#pragma unroll
    for (int t = 0; t < 4; ++t)
#pragma unroll
      for (int r = 0; r < 8; ++r) acc[t][r] *= alpha;

    // Build P^T B-fragment: pack to f16x2, swap missing halves via lane^16.
    unsigned p0[4], p1[4];
#pragma unroll
    for (int j = 0; j < 4; ++j) {
      p0[j] = __builtin_bit_cast(
          unsigned, __builtin_amdgcn_cvt_pkrtz(st0[2 * j], st0[2 * j + 1]));
      p1[j] = __builtin_bit_cast(
          unsigned, __builtin_amdgcn_cvt_pkrtz(st1[2 * j], st1[2 * j + 1]));
    }
    v8u pw;
#pragma unroll
    for (int j = 0; j < 4; ++j) {
      const unsigned x0 = xhalf_u(p0[j]);
      const unsigned x1 = xhalf_u(p1[j]);
      pw[j]     = hb ? x1 : p0[j];    // elements 0..7:  keys (hb?16:0)+0..7
      pw[4 + j] = hb ? p1[j] : x0;    // elements 8..15: keys (hb?24:8)+0..7
    }
    const v16h pb = __builtin_bit_cast(v16h, pw);

    // O^T += V^T * P^T: A-frags straight from the transposed V buffer.
#pragma unroll
    for (int t = 0; t < 4; ++t) {
      const _Float16* vp = vrow[t] + kv + hb * 16;
      v16h vf = combine8(*(const half8*)vp, *(const half8*)(vp + 8));
      acc[t] = wmma_f16(vf, pb, acc[t]);
    }
  }

  // Epilogue: lane owns query q; tile t rows are dk = t*16 + 8*hb + r.
  const float invl = 1.0f / l;
  float* ob = attn + ((size_t)b * S_ + q) * D_ + h * DK_;
#pragma unroll
  for (int t = 0; t < 4; ++t) {
    float4 v0, v1;
    v0.x = acc[t][0] * invl; v0.y = acc[t][1] * invl;
    v0.z = acc[t][2] * invl; v0.w = acc[t][3] * invl;
    v1.x = acc[t][4] * invl; v1.y = acc[t][5] * invl;
    v1.z = acc[t][6] * invl; v1.w = acc[t][7] * invl;
    *(float4*)(ob + t * 16 + 8 * hb)     = v0;
    *(float4*)(ob + t * 16 + 8 * hb + 4) = v1;
  }
}

// ---------------------------------------------------------------------------
// Host launcher.
// ---------------------------------------------------------------------------
extern "C" void kernel_launch(void* const* d_in, const int* in_sizes, int n_in,
                              void* d_out, int out_size, void* d_ws, size_t ws_size,
                              hipStream_t stream) {
  (void)in_sizes; (void)n_in; (void)out_size; (void)ws_size;
  const float* x       = (const float*)d_in[0];
  const float* ln_in_g = (const float*)d_in[2];
  const float* ln_in_b = (const float*)d_in[3];
  const float* Wq      = (const float*)d_in[4];
  const float* bq      = (const float*)d_in[5];
  const float* Wk      = (const float*)d_in[6];
  const float* bk      = (const float*)d_in[7];
  const float* Wv      = (const float*)d_in[8];
  const float* bv      = (const float*)d_in[9];
  const float* ln1_g   = (const float*)d_in[10];
  const float* ln1_b   = (const float*)d_in[11];
  const float* W1      = (const float*)d_in[12];
  const float* b1      = (const float*)d_in[13];
  const float* W2      = (const float*)d_in[14];
  const float* b2      = (const float*)d_in[15];
  const float* ln2_g   = (const float*)d_in[16];
  const float* ln2_b   = (const float*)d_in[17];

  char* ws = (char*)d_ws;
  size_t off = 0;
  auto alloc = [&](size_t bytes) {
    char* p = ws + off;
    off += (bytes + 255) & ~(size_t)255;
    return p;
  };
  float*    xn32   = (float*)   alloc((size_t)ROWS_ * D_ * 4);
  _Float16* xn16   = (_Float16*)alloc((size_t)ROWS_ * D_ * 2);
  _Float16* w3T    = (_Float16*)alloc((size_t)QKVL_ * D_ * 2);  // [1536,512]
  float*    bias3  = (float*)   alloc((size_t)QKVL_ * 4);
  _Float16* w1T    = (_Float16*)alloc((size_t)DFF_ * D_ * 2);   // [2048,512]
  _Float16* w2T    = (_Float16*)alloc((size_t)D_ * DFF_ * 2);   // [512,2048]
  _Float16* qkv16  = (_Float16*)alloc((size_t)ROWS_ * QKVL_ * 2);
  _Float16* vtg    = (_Float16*)alloc((size_t)B_ * H_ * DK_ * S_ * 2);
  float*    attn32 = (float*)   alloc((size_t)ROWS_ * D_ * 4);
  float*    ao32   = (float*)   alloc((size_t)ROWS_ * D_ * 4);  // attn_out
  _Float16* o16    = (_Float16*)alloc((size_t)ROWS_ * D_ * 2);  // LN1 out
  _Float16* h16    = (_Float16*)alloc((size_t)ROWS_ * DFF_ * 2);
  float*    o232   = (float*)   alloc((size_t)ROWS_ * D_ * 4);

  // Weight conversion, pre-transposed to [N,K].
  const int nW = (D_ * D_) / 4;         // float4 units
  cvt_f32_f16_T<<<nW / 256, 256, 0, stream>>>(Wq, w3T + (size_t)0 * D_ * D_,
                                              nW, D_ / 4, D_);
  cvt_f32_f16_T<<<nW / 256, 256, 0, stream>>>(Wk, w3T + (size_t)1 * D_ * D_,
                                              nW, D_ / 4, D_);
  cvt_f32_f16_T<<<nW / 256, 256, 0, stream>>>(Wv, w3T + (size_t)2 * D_ * D_,
                                              nW, D_ / 4, D_);
  const int nW1 = (D_ * DFF_) / 4;
  cvt_f32_f16_T<<<nW1 / 256, 256, 0, stream>>>(W1, w1T, nW1, DFF_ / 4, D_);
  cvt_f32_f16_T<<<nW1 / 256, 256, 0, stream>>>(W2, w2T, nW1, D_ / 4, DFF_);
  pack_bias3<<<QKVL_ / 256, 256, 0, stream>>>(bq, bk, bv, bias3);

  // xn = LN(x): f32 for residual + f16 for GEMM.
  ln_kernel<0><<<ROWS_, 128, 0, stream>>>(x, nullptr, ln_in_g, ln_in_b,
                                          nullptr, xn32, xn16);

  // Fused QKV projection: [4096,512] x [512,1536] (V routed to vtg).
  gemm_wmma<2><<<(ROWS_ / 128) * (QKVL_ / 128), 256, 0, stream>>>(
      xn16, w3T, bias3, qkv16, nullptr, nullptr, vtg, ROWS_, QKVL_, D_);

  // Windowed attention (LDS-free).
  attn_wmma<<<B_ * H_ * (S_ / CHUNK_), 256, 0, stream>>>(qkv16, vtg, attn32);

  // attn_out = attn + xn (saved f32); o = LN1(attn_out) -> f16.
  ln_kernel<1><<<ROWS_, 128, 0, stream>>>(attn32, xn32, ln1_g, ln1_b,
                                          ao32, nullptr, o16);

  // FFN: h = o @ W1 + b1 ; o2 = h @ W2 + b2 + attn_out.
  gemm_wmma<0><<<(ROWS_ / 128) * (DFF_ / 128), 256, 0, stream>>>(
      o16, w1T, b1, h16, nullptr, nullptr, nullptr, ROWS_, DFF_, D_);
  gemm_wmma<1><<<(ROWS_ / 128) * (D_ / 128), 256, 0, stream>>>(
      h16, w2T, b2, nullptr, o232, ao32, nullptr, ROWS_, D_, DFF_);

  // out = LN2(o2).
  ln_kernel<2><<<ROWS_, 128, 0, stream>>>(o232, nullptr, ln2_g, ln2_b,
                                          nullptr, (float*)d_out, nullptr);
}
